// YOLOV8LabelEncoder_85461259256075
// MI455X (gfx1250) — matrix operand order repaired and checked by reading, not compiled, not run on hardware
//
#include <hip/hip_runtime.h>
#include <hip/hip_bf16.h>
#include <math.h>

// Problem constants (match reference setup_inputs)
#define BB 16
#define NBB 64
#define NCC 80
#define TOPK 10
#define KEPSF 1e-7f
#define EPSF 1e-9f
#define INV_PI2 0.4052847345693511f   // 4/pi^2

typedef __attribute__((ext_vector_type(16))) _Float16 v16h;
typedef __attribute__((ext_vector_type(8)))  float    v8f;

// ---------------------------------------------------------------------------
// K0: sc[b,g,a] = pd_scores[b,a,label[b,g]] as a WMMA GEMM:
//     OneHot(labels)[64 x 96] @ pd_scores^T[96 x 8400] per batch.
// One wave computes a 16(gt) x 16(anchor) tile with 3 chained
// v_wmma_f32_16x16x32_f16 ops (classes padded 80 -> 96).
// ---------------------------------------------------------------------------
__global__ __launch_bounds__(32)
void k0_score_gemm(const float* __restrict__ pd_scores,
                   const float* __restrict__ gt_labels,
                   float* __restrict__ sc, int NA)
{
    const int lane = threadIdx.x & 31;
    const int half = lane >> 4;      // 0: lanes 0-15, 1: lanes 16-31
    const int l15  = lane & 15;
    const int a0   = blockIdx.x * 16;
    const int g0   = blockIdx.y * 16;
    const int b    = blockIdx.z;

    // A-matrix row (both lane halves hold M=0..15, different K ranges)
    const int   m     = g0 + l15;
    const float labf  = gt_labels[b * NBB + m];
    int lbl = (int)labf; if (lbl < 0) lbl = 0;
    const bool lab_ok = (labf >= 0.0f);

    // B-matrix column (anchor)
    const int n = a0 + l15;
    const float* srow = pd_scores + ((size_t)b * NA + n) * NCC;

    v8f acc = {};
    #pragma unroll
    for (int ks = 0; ks < 3; ++ks) {
        const int cb = ks * 32;

        // A fragment: one-hot of label. K order matches B:
        //   lane<16 : K = i
        //   lane>=16: K = i+16
        v16h afrag;
        #pragma unroll
        for (int i = 0; i < 16; ++i) {
            const int K = half * 16 + i;
            const int c = cb + K;
            afrag[i] = (lab_ok && c == lbl) ? (_Float16)1.0f : (_Float16)0.0f;
        }

        // B fragment: B[K,n] = pd_scores[b,n,cb+K].
        //   lanes 0-15: K=0..15, lanes 16-31: K=16..31 (i -> K-offset i)
        v16h bfrag;
        const int cbase = cb + half * 16;
        if (cbase < NCC) {  // 16 contiguous, 16B-aligned floats
            const float4* p = reinterpret_cast<const float4*>(srow + cbase);
            #pragma unroll
            for (int q = 0; q < 4; ++q) {
                const float4 v = p[q];
                bfrag[4*q+0] = (_Float16)v.x;
                bfrag[4*q+1] = (_Float16)v.y;
                bfrag[4*q+2] = (_Float16)v.z;
                bfrag[4*q+3] = (_Float16)v.w;
            }
        } else {
            #pragma unroll
            for (int i = 0; i < 16; ++i) bfrag[i] = (_Float16)0.0f;
        }

        acc = __builtin_amdgcn_wmma_f32_16x16x32_f16(
            false, afrag, false, bfrag, (short)0, acc, false, false);
    }

    // D layout: VGPR r -> M = r + 8*half, N = lane&15
    #pragma unroll
    for (int r = 0; r < 8; ++r) {
        const int g = g0 + r + half * 8;
        sc[((size_t)b * NBB + g) * NA + n] = acc[r];
    }
}

// ---------------------------------------------------------------------------
// K1: per (b,g,a): CIoU, in-gt mask, align metric (in place over sc buffer).
// ---------------------------------------------------------------------------
__global__ __launch_bounds__(256)
void k1_metric(const float* __restrict__ pd_bboxes,
               const float* __restrict__ anc_points,
               const float* __restrict__ gt_bboxes,
               const unsigned char* __restrict__ mask_gt,
               float* __restrict__ scmet,          // in: sc, out: metric
               float* __restrict__ ovl,
               unsigned char* __restrict__ validb,
               unsigned char* __restrict__ maskpos,
               int NA)
{
    const int a = blockIdx.x * blockDim.x + threadIdx.x;
    if (a >= NA) return;
    const int g = blockIdx.y, b = blockIdx.z;
    const size_t bg  = (size_t)b * NBB + g;
    const size_t idx = bg * NA + a;

    const float x1a = gt_bboxes[bg*4+0], y1a = gt_bboxes[bg*4+1];
    const float x2a = gt_bboxes[bg*4+2], y2a = gt_bboxes[bg*4+3];
    const unsigned char mg = mask_gt[bg];

    const float ax = anc_points[a*2+0], ay = anc_points[a*2+1];
    const float dmin = fminf(fminf(ax - x1a, ay - y1a),
                             fminf(x2a - ax, y2a - ay));
    const bool valid = (dmin > EPSF) && (mg != 0);

    const size_t pb = ((size_t)b * NA + a) * 4;
    const float x1b = pd_bboxes[pb+0], y1b = pd_bboxes[pb+1];
    const float x2b = pd_bboxes[pb+2], y2b = pd_bboxes[pb+3];

    // CIoU (gt vs pred), matching the reference formula
    const float w1 = x2a - x1a, h1 = y2a - y1a + KEPSF;
    const float w2 = x2b - x1b, h2 = y2b - y1b + KEPSF;
    const float iw = fminf(x2a, x2b) - fmaxf(x1a, x1b);
    const float ih = fminf(y2a, y2b) - fmaxf(y1a, y1b);
    const float inter = fmaxf(iw, 0.f) * fmaxf(ih, 0.f);
    const float uni = w1 * h1 + w2 * h2 - inter + KEPSF;
    const float iou = inter / uni;
    const float cw = fmaxf(x2a, x2b) - fminf(x1a, x1b);
    const float ch = fmaxf(y2a, y2b) - fminf(y1a, y1b);
    const float c2 = cw * cw + ch * ch + KEPSF;
    const float dx = 0.5f * (x1a + x2a) - 0.5f * (x1b + x2b);
    const float dy = 0.5f * (y1a + y2a) - 0.5f * (y1b + y2b);
    const float rho2 = dx * dx + dy * dy;
    const float dat = atanf(w2 / h2) - atanf(w1 / h1);
    const float v = INV_PI2 * dat * dat;
    const float alpha = v / (v - iou + 1.0f + KEPSF);
    const float ciou = iou - (rho2 / c2 + v * alpha);

    const float o = valid ? fmaxf(ciou, 0.0f) : 0.0f;
    float s = valid ? scmet[idx] : 0.0f;
    s = fmaxf(s, 0.0f);
    const float o2 = o * o;
    const float o6 = o2 * o2 * o2;

    scmet[idx]  = sqrtf(s) * o6;      // align_metric (alpha=0.5, beta=6)
    ovl[idx]    = o;
    validb[idx] = valid ? 1 : 0;
    maskpos[idx]= 0;
}

// ---------------------------------------------------------------------------
// K2: top-10 per (b,g) by iterative block max; smallest-index tie break.
// ---------------------------------------------------------------------------
__global__ __launch_bounds__(256)
void k2_topk(const float* __restrict__ metric,
             const unsigned char* __restrict__ validb,
             const unsigned char* __restrict__ mask_gt,
             unsigned char* __restrict__ maskpos, int NA)
{
    __shared__ float sv[256];
    __shared__ int   si[256];
    __shared__ int   chosen[TOPK];
    const int g = blockIdx.x, b = blockIdx.y, t = threadIdx.x;
    const size_t base = ((size_t)b * NBB + g) * NA;

    for (int k = 0; k < TOPK; ++k) {
        float bestv = -1.0f; int besti = NA;
        for (int a = t; a < NA; a += 256) {
            bool skip = false;
            for (int j = 0; j < k; ++j)
                if (chosen[j] == a) { skip = true; break; }
            if (skip) continue;
            const float v = metric[base + a];
            if (v > bestv || (v == bestv && a < besti)) { bestv = v; besti = a; }
        }
        sv[t] = bestv; si[t] = besti;
        __syncthreads();
        for (int s = 128; s > 0; s >>= 1) {
            if (t < s) {
                if (sv[t+s] > sv[t] || (sv[t+s] == sv[t] && si[t+s] < si[t])) {
                    sv[t] = sv[t+s]; si[t] = si[t+s];
                }
            }
            __syncthreads();
        }
        if (t == 0) chosen[k] = si[0];
        __syncthreads();
    }
    if (t < TOPK) {
        const int a = chosen[t];
        if (mask_gt[b * NBB + g] && a < NA)
            maskpos[base + a] = validb[base + a] ? 1 : 0;
    }
}

// ---------------------------------------------------------------------------
// K3: per (b,a): fg count, conflict resolution via argmax overlap, tgt idx.
// ---------------------------------------------------------------------------
__global__ __launch_bounds__(256)
void k3_resolve(const float* __restrict__ ovl,
                unsigned char* __restrict__ maskpos,
                int* __restrict__ tgi,
                float* __restrict__ out_fg, int NA)
{
    const size_t tid = (size_t)blockIdx.x * blockDim.x + threadIdx.x;
    if (tid >= (size_t)BB * NA) return;
    const int b = (int)(tid / NA), a = (int)(tid % NA);
    const size_t base = (size_t)b * NBB * NA + a;

    int cnt = 0, firstg = 0; bool seen = false;
    float bo = -1.0f; int bog = 0;
    for (int g = 0; g < NBB; ++g) {
        const size_t idx = base + (size_t)g * NA;
        if (maskpos[idx]) { cnt++; if (!seen) { firstg = g; seen = true; } }
        const float o = ovl[idx];
        if (o > bo) { bo = o; bog = g; }   // first max (argmax semantics)
    }
    float fg; int tg;
    if (cnt > 1) {
        for (int g = 0; g < NBB; ++g)
            maskpos[base + (size_t)g * NA] = (g == bog) ? 1 : 0;
        fg = 1.0f; tg = bog;
    } else {
        fg = (float)cnt; tg = seen ? firstg : 0;
    }
    out_fg[(size_t)b * NA + a] = fg;
    tgi[(size_t)b * NA + a] = tg;
}

// ---------------------------------------------------------------------------
// K4: per (b,g): ratio = max(ovl*mask) / (max(metric*mask) + eps)
// ---------------------------------------------------------------------------
__global__ __launch_bounds__(256)
void k4_ratio(const float* __restrict__ metric,
              const float* __restrict__ ovl,
              const unsigned char* __restrict__ maskpos,
              float* __restrict__ ratio, int NA)
{
    __shared__ float sa[256], so[256];
    const int g = blockIdx.x, b = blockIdx.y, t = threadIdx.x;
    const size_t base = ((size_t)b * NBB + g) * NA;
    float ma = 0.0f, mo = 0.0f;
    for (int a = t; a < NA; a += 256) {
        const float m = maskpos[base + a] ? 1.0f : 0.0f;
        ma = fmaxf(ma, metric[base + a] * m);
        mo = fmaxf(mo, ovl[base + a] * m);
    }
    sa[t] = ma; so[t] = mo;
    __syncthreads();
    for (int s = 128; s > 0; s >>= 1) {
        if (t < s) { sa[t] = fmaxf(sa[t], sa[t+s]); so[t] = fmaxf(so[t], so[t+s]); }
        __syncthreads();
    }
    if (t == 0) ratio[b * NBB + g] = so[0] / (sa[0] + EPSF);
}

// ---------------------------------------------------------------------------
// K5: per (b,a): norm, gather target bbox/label, dense one-hot score write.
// ---------------------------------------------------------------------------
__global__ __launch_bounds__(256)
void k5_write(const float* __restrict__ metric,
              const unsigned char* __restrict__ maskpos,
              const float* __restrict__ ratio,
              const int* __restrict__ tgi,
              const float* __restrict__ out_fg,
              const float* __restrict__ gt_labels,
              const float* __restrict__ gt_bboxes,
              float* __restrict__ out_bbox,
              float* __restrict__ out_scr, int NA)
{
    const size_t tid = (size_t)blockIdx.x * blockDim.x + threadIdx.x;
    if (tid >= (size_t)BB * NA) return;
    const int b = (int)(tid / NA), a = (int)(tid % NA);

    float norm = 0.0f;
    const size_t col = (size_t)b * NBB * NA + a;
    for (int g = 0; g < NBB; ++g) {
        const size_t idx = col + (size_t)g * NA;
        if (maskpos[idx])
            norm = fmaxf(norm, metric[idx] * ratio[b * NBB + g]);
    }
    const int tg = tgi[(size_t)b * NA + a];
    const float fg = out_fg[(size_t)b * NA + a];
    int lbl = (int)gt_labels[b * NBB + tg]; if (lbl < 0) lbl = 0;

    const float* gb = gt_bboxes + ((size_t)b * NBB + tg) * 4;
    float* ob = out_bbox + ((size_t)b * NA + a) * 4;
    ob[0] = gb[0]; ob[1] = gb[1]; ob[2] = gb[2]; ob[3] = gb[3];

    const float val = (fg > 0.0f) ? norm : 0.0f;
    float4* row = reinterpret_cast<float4*>(out_scr + ((size_t)b * NA + a) * NCC);
    #pragma unroll
    for (int q = 0; q < NCC / 4; ++q) row[q] = make_float4(0.f, 0.f, 0.f, 0.f);
    out_scr[((size_t)b * NA + a) * NCC + lbl] = val;
}

// ---------------------------------------------------------------------------
extern "C" void kernel_launch(void* const* d_in, const int* in_sizes, int n_in,
                              void* d_out, int out_size, void* d_ws, size_t ws_size,
                              hipStream_t stream)
{
    const float* pd_scores  = (const float*)d_in[0];
    const float* pd_bboxes  = (const float*)d_in[1];
    const float* anc_points = (const float*)d_in[2];
    const float* gt_labels  = (const float*)d_in[3];
    const float* gt_bboxes  = (const float*)d_in[4];
    const unsigned char* mask_gt = (const unsigned char*)d_in[5]; // jnp bool -> 1B

    const int NA = in_sizes[2] / 2;                 // 8400
    const size_t BGA = (size_t)BB * NBB * NA;

    // Workspace layout (~86.6 MB; all L2-resident on MI455X's 192 MB L2)
    float* scmet = (float*)d_ws;                    // B*NB*NA (sc -> metric)
    float* ovl   = scmet + BGA;                     // B*NB*NA
    float* ratio = ovl + BGA;                       // B*NB
    int*   tgi   = (int*)(ratio + BB * NBB);        // B*NA
    unsigned char* maskpos = (unsigned char*)(tgi + (size_t)BB * NA); // B*NB*NA
    unsigned char* validb  = maskpos + BGA;         // B*NB*NA

    float* out_bbox = (float*)d_out;                        // B*NA*4
    float* out_scr  = out_bbox + (size_t)BB * NA * 4;       // B*NA*80
    float* out_fg   = out_scr + (size_t)BB * NA * NCC;      // B*NA

    dim3 g0(NA / 16, NBB / 16, BB);
    k0_score_gemm<<<g0, 32, 0, stream>>>(pd_scores, gt_labels, scmet, NA);

    dim3 g1((NA + 255) / 256, NBB, BB);
    k1_metric<<<g1, 256, 0, stream>>>(pd_bboxes, anc_points, gt_bboxes, mask_gt,
                                      scmet, ovl, validb, maskpos, NA);

    dim3 g2(NBB, BB);
    k2_topk<<<g2, 256, 0, stream>>>(scmet, validb, mask_gt, maskpos, NA);

    const int nba = BB * NA;
    k3_resolve<<<(nba + 255) / 256, 256, 0, stream>>>(ovl, maskpos, tgi, out_fg, NA);

    k4_ratio<<<g2, 256, 0, stream>>>(scmet, ovl, maskpos, ratio, NA);

    k5_write<<<(nba + 255) / 256, 256, 0, stream>>>(scmet, maskpos, ratio, tgi,
                                                    out_fg, gt_labels, gt_bboxes,
                                                    out_bbox, out_scr, NA);
}
